// GraphFusionModel_36653250904481
// MI455X (gfx1250) — matrix operand order repaired and checked by reading, not compiled, not run on hardware
//
#include <hip/hip_runtime.h>
#include <math.h>
#include <stdint.h>

// ---------------------------------------------------------------------------
// GraphFusionModel for MI455X (gfx1250, wave32, WMMA + TDM).
//  - Edge MLP (dominant 5.4 GFLOP), attention QK^T / P*V, and all dense GEMMs
//    run on v_wmma_f32_16x16x32_f16 (f16 operands, f32 accumulation).
//  - GEMM operand tiles are staged into LDS by the Tensor Data Mover
//    (tensor_load_to_lds + s_wait_tensorcnt), per CDNA5 ISA.
//    This toolchain exposes the 6-arg builtin (g0, g1, g2, g3, g_ext, cpol).
//  - Irregular parts (KNN top-8, softmax, LN, pooling) use wave32 shuffles.
//  - Fragment builds are branchless (clamped unconditional loads + cndmask)
//    to avoid per-element saveexec around global loads.
// ---------------------------------------------------------------------------

#define BB     8
#define LL     512
#define DM     128
#define NH     4
#define DK     32
#define DFF    256
#define TLEN   20
#define KNN    8
#define IN_DIM 19

typedef __attribute__((ext_vector_type(16))) _Float16     v16h;
typedef __attribute__((ext_vector_type(8)))  float        v8f;
typedef __attribute__((ext_vector_type(4)))  unsigned int v4u;
typedef __attribute__((ext_vector_type(8)))  int          v8i;
typedef __attribute__((ext_vector_type(4)))  int          v4i;

union AFrag { v16h v; _Float16 h[16]; };
union CFrag { v8f  v; float    f[8];  };
union G0    { v4u  v; unsigned int d[4]; };
union G1    { v8i  v; int d[8]; };

// ---------------------------------------------------------------------------
// TDM: 2-D tile load Global -> LDS.  Descriptor packing per ISA 8.3/8.4:
//   group0: [1:0]=count=1, [63:32]=lds_addr, [120:64]=global_addr, [127:126]=2
//   group1: data_size(4B)=2<<16 ; tensor_dim0[79:48] ; tensor_dim1[111:80] ;
//           tile_dim0[127:112] ; tile_dim1[143:128] ; dim0_stride[207:160]
// One wave per block => exactly one TDM issue per call (EXEC-independent).
// ---------------------------------------------------------------------------
__device__ __forceinline__ void tdm_load_2d(unsigned int lds_addr, const void* gptr,
                                            unsigned int tile_x, unsigned int tile_y,
                                            unsigned long long stride0,
                                            unsigned long long dim0,
                                            unsigned long long dim1) {
  unsigned long long ga = (unsigned long long)(uintptr_t)gptr;
  G0 g0;
  g0.d[0] = 1u;                                             // count=1 (user D#)
  g0.d[1] = lds_addr;                                       // LDS byte address
  g0.d[2] = (unsigned int)(ga & 0xffffffffull);
  g0.d[3] = (unsigned int)((ga >> 32) & 0x01ffffffull) | (2u << 30);  // type=2
  G1 g1;
  g1.d[0] = (int)(2u << 16);                                // data_size = 4B
  g1.d[1] = (int)((dim0 & 0xffffull) << 16);
  g1.d[2] = (int)(((dim0 >> 16) & 0xffffull) | ((dim1 & 0xffffull) << 16));
  g1.d[3] = (int)(((dim1 >> 16) & 0xffffull) | ((unsigned long long)(tile_x & 0xffffu) << 16));
  g1.d[4] = (int)(tile_y & 0xffffu);                        // tile_dim1 (tile_dim2=0)
  g1.d[5] = (int)(stride0 & 0xffffffffull);
  g1.d[6] = (int)((stride0 >> 32) & 0xffffull);
  g1.d[7] = 0;
  v4i z4 = {0, 0, 0, 0};
  v8i z8 = {0, 0, 0, 0, 0, 0, 0, 0};
  __builtin_amdgcn_tensor_load_to_lds(g0.v, g1.v, z4, z4, z8, 0);
}

// ---------------------------------------------------------------------------
// 1. Encoder: h = feat(19) @ enc_w + enc_b (K=19 too ragged for WMMA).
// ---------------------------------------------------------------------------
__global__ void k_encode(const float* __restrict__ feat, const float* __restrict__ w,
                         const float* __restrict__ b, float* __restrict__ h) {
  int n = blockIdx.x;
  int d = threadIdx.x;
  const float* f = feat + (size_t)n * IN_DIM;
  float acc = b[d];
#pragma unroll
  for (int i = 0; i < IN_DIM; i++) acc += f[i] * w[i * DM + d];
  h[(size_t)n * DM + d] = acc;
}

// ---------------------------------------------------------------------------
// 2. Edge MLP on WMMA.  One wave per (b, q, 16-wide k-tile); pair index is the
//    WMMA M dimension.  K=6 pads into one 16x16x32 step: per ISA layouts,
//    lanes>=16 and elements with k>=6 are structurally zero, so fragments
//    build with constant indices.  Stage1: 8 WMMAs (6->128) + bias + ReLU
//    -> LDS f16 tile; Stage2: 4 WMMAs (128->4, N padded to 16).
//    Also emits the KNN metric.  All fragment loads are unconditional
//    (indices clamped in-bounds); lane predicates are register selects.
// ---------------------------------------------------------------------------
__global__ void k_edge_wmma(const float* __restrict__ feat, const int* __restrict__ node_t,
                            const int* __restrict__ node_m, const float* __restrict__ mask,
                            const float* __restrict__ w1, const float* __restrict__ b1,
                            const float* __restrict__ w2, const float* __restrict__ b2,
                            float* __restrict__ bias, float* __restrict__ metric) {
  __shared__ _Float16 sH[16 * DM];            // hid tile, 4 KB
  int b = blockIdx.z, q = blockIdx.y;
  int k0 = blockIdx.x * 16;
  int lane = threadIdx.x, q15 = lane & 15;
  int hi = (lane >= 16) ? 1 : 0;
  const _Float16 hz = (_Float16)0.f;

  // ---- per-pair features (pair = (q, k0+q15); both lane halves duplicate) --
  int iq = b * LL + q, ik = b * LL + k0 + q15;
  const float* fq = feat + (size_t)iq * IN_DIM;
  const float* fk = feat + (size_t)ik * IN_DIM;
  int ti = max(node_t[iq], 0), tj = max(node_t[ik], 0);
  int mi = max(node_m[iq], 0), mj = max(node_m[ik], 0);
  float dt        = fabsf((float)(ti - tj)) * (1.0f / (float)(TLEN - 1));
  float same_mod  = (mi == mj) ? 1.f : 0.f;
  float same_time = (ti == tj) ? 1.f : 0.f;
  float dx = fq[0] - fk[0], dy = fq[1] - fk[1], dz = fq[2] - fk[2];
  float d2 = dx * dx + dy * dy + dz * dz;
  float dist = (d2 > 0.f) ? sqrtf(d2) : 0.f;
  bool  pv   = (fq[9] > 0.5f) && (fk[9] > 0.5f);
  if (!pv) dist = 50.f;
  float dconf = fabsf(fq[7] - fk[7]);
  bool  valid = (mask[iq] > 0.f) && (mask[ik] > 0.f);
  float vf = valid ? 1.f : 0.f;
  float ef0 = dist * vf, ef1 = dt * vf, ef2 = same_mod * vf;
  float ef3 = same_time * vf, ef4 = dconf * vf, ef5 = (pv ? 1.f : 0.f) * vf;

  if (lane < 16) {                            // KNN metric, one lane per pair
    float m = valid ? (dist + dt) : 1e9f;
    if (q == (k0 + q15) && mask[iq] > 0.f) m = 0.f;
    metric[((size_t)b * LL + q) * LL + k0 + q15] = m;
  }

  // ---- stage 1: A = edge features (16 pairs x K=32, k>=6 zero) -------------
  AFrag a;
#pragma unroll
  for (int e = 0; e < 16; e++) a.h[e] = hz;
  a.h[0] = hi ? hz : (_Float16)ef0;  a.h[1] = hi ? hz : (_Float16)ef1;
  a.h[2] = hi ? hz : (_Float16)ef2;  a.h[3] = hi ? hz : (_Float16)ef3;
  a.h[4] = hi ? hz : (_Float16)ef4;  a.h[5] = hi ? hz : (_Float16)ef5;

#pragma unroll
  for (int nt = 0; nt < DM / 16; nt++) {      // 8 WMMAs
    int n = nt * 16 + q15;
    AFrag bf;
#pragma unroll
    for (int e = 0; e < 16; e++) {
      if (e < 6) {                            // compile-time after unroll
        float wv = w1[e * DM + n];            // unconditional, in-bounds
        bf.h[e] = hi ? hz : (_Float16)wv;     // lane select, no exec churn
      } else {
        bf.h[e] = hz;
      }
    }
    CFrag c;
#pragma unroll
    for (int i = 0; i < 8; i++) c.f[i] = 0.f;
    c.v = __builtin_amdgcn_wmma_f32_16x16x32_f16(false, a.v, false, bf.v,
                                                 (short)0, c.v, false, false);
#pragma unroll
    for (int r = 0; r < 8; r++) {             // bias + ReLU -> LDS f16
      int m = r + hi * 8;
      float v = fmaxf(c.f[r] + b1[n], 0.f);
      sH[m * DM + n] = (_Float16)v;
    }
  }
  __syncthreads();

  // ---- stage 2: bias_out = hid(16x128) @ W2(128x4), N padded to 16 ---------
  int q3 = q15 & 3;                           // clamped column (in-bounds)
  bool hvalid = (q15 < 4);
  CFrag c2;
#pragma unroll
  for (int i = 0; i < 8; i++) c2.f[i] = 0.f;
#pragma unroll
  for (int ks = 0; ks < DM / 32; ks++) {      // 4 WMMAs
    AFrag ah, bf;
#pragma unroll
    for (int e = 0; e < 16; e++) {
      int ka = ks * 32 + ((e < 8) ? e : e + 8) + hi * 8;
      ah.h[e] = sH[q15 * DM + ka];
      int kb = ks * 32 + ((e < 8) ? e : e + 8) + hi * 8;
      float wv = w2[kb * 4 + q3];             // unconditional, in-bounds
      bf.h[e] = hvalid ? (_Float16)wv : hz;
    }
    c2.v = __builtin_amdgcn_wmma_f32_16x16x32_f16(false, ah.v, false, bf.v,
                                                  (short)0, c2.v, false, false);
  }
  if (hvalid) {
    size_t hs = (size_t)LL * LL;
#pragma unroll
    for (int r = 0; r < 8; r++) {
      int m = r + hi * 8;
      bias[((size_t)(b * NH + q15)) * hs + (size_t)q * LL + k0 + m] =
          c2.f[r] + b2[q15];
    }
  }
}

// ---------------------------------------------------------------------------
// 3. KNN: one wave per row, 8x stable argmin via wave32 shuffle butterfly.
// ---------------------------------------------------------------------------
__global__ void k_knn(const float* __restrict__ metric, const float* __restrict__ mask,
                      unsigned char* __restrict__ sparse) {
  __shared__ float row[LL];
  int b = blockIdx.y, q = blockIdx.x;
  int lane = threadIdx.x;
  const float* mrow = metric + ((size_t)b * LL + q) * LL;
  for (int j = lane; j < LL; j += 32) row[j] = mrow[j];
  __syncthreads();
  float mq = mask[b * LL + q];
  for (int it = 0; it < KNN; ++it) {
    float mval = 3e38f; int midx = LL;
    for (int j = lane; j < LL; j += 32) {
      float v = row[j];
      if (v < mval || (v == mval && j < midx)) { mval = v; midx = j; }
    }
    for (int off = 16; off > 0; off >>= 1) {
      float ov = __shfl_xor(mval, off, 32);
      int   oi = __shfl_xor(midx, off, 32);
      if (ov < mval || (ov == mval && oi < midx)) { mval = ov; midx = oi; }
    }
    if (lane == 0) {
      row[midx] = 3e38f;
      if (mq > 0.f && mask[b * LL + midx] > 0.f) {
        sparse[((size_t)b * LL + q) * LL + midx] = 1;
        sparse[((size_t)b * LL + midx) * LL + q] = 1;
      }
    }
    __syncthreads();
  }
  if (lane == 0 && mq > 0.f) sparse[((size_t)b * LL + q) * LL + q] = 1;
}

// ---------------------------------------------------------------------------
// 4. WMMA GEMM with TDM-staged operands: C[M,N] = A[M,K] @ W[K,N] + bias.
//    One wave per 16x16 C tile; each K-step DMA-loads a 16x32 A tile and a
//    32x16 W tile into LDS (tensor_load_to_lds), waits on TENSORcnt, then
//    feeds the matrix pipe.
// ---------------------------------------------------------------------------
__global__ void k_gemm_wmma(const float* __restrict__ A, const float* __restrict__ W,
                            const float* __restrict__ bias, float* __restrict__ C,
                            int M, int N, int K, int relu) {
  __shared__ float sA[16 * 32];               // 2 KB
  __shared__ float sB[32 * 16];               // 2 KB
  int lane = threadIdx.x;
  int n0 = blockIdx.x * 16, m0 = blockIdx.y * 16;
  int hi = (lane >= 16) ? 1 : 0;
  int q15 = lane & 15;
  unsigned int ldsA = (unsigned int)(uintptr_t)&sA[0];
  unsigned int ldsB = (unsigned int)(uintptr_t)&sB[0];

  CFrag c;
#pragma unroll
  for (int i = 0; i < 8; i++) c.f[i] = 0.f;

  for (int k0 = 0; k0 < K; k0 += 32) {
    // A tile: rows m0..m0+15, cols k0..k0+31 (x = K contiguous)
    tdm_load_2d(ldsA, &A[(size_t)m0 * K + k0], 32, 16,
                (unsigned long long)K, (unsigned long long)K, (unsigned long long)M);
    // W tile: rows k0..k0+31, cols n0..n0+15 (x = N contiguous)
    tdm_load_2d(ldsB, &W[(size_t)k0 * N + n0], 16, 32,
                (unsigned long long)N, (unsigned long long)N, (unsigned long long)K);
    __builtin_amdgcn_s_wait_tensorcnt(0);
    __syncthreads();

    AFrag a, bf;
#pragma unroll
    for (int e = 0; e < 16; e++) {
      int ka = ((e < 8) ? e : e + 8) + hi * 8;
      a.h[e]  = (_Float16)sA[q15 * 32 + ka];
      int kb = ((e < 8) ? e : e + 8) + hi * 8;
      bf.h[e] = (_Float16)sB[kb * 16 + q15];
    }
    c.v = __builtin_amdgcn_wmma_f32_16x16x32_f16(false, a.v, false, bf.v,
                                                 (short)0, c.v, false, false);
    __syncthreads();                          // fragments read before next DMA
  }
#pragma unroll
  for (int r = 0; r < 8; r++) {
    int m = m0 + r + hi * 8;
    float v = c.f[r] + bias[n0 + q15];
    if (relu) v = fmaxf(v, 0.f);
    C[(size_t)m * N + (n0 + q15)] = v;
  }
}

// ---------------------------------------------------------------------------
// 5. Attention on WMMA (flash-style, one wave per 16-query tile per (b,h)).
//    QK^T: 32 WMMAs (Q fragment hoisted); epilogue scale+bias+masks into a
//    16x512 LDS score buffer; wave-cooperative softmax; P*V: 2x16 WMMAs.
// ---------------------------------------------------------------------------
__global__ void k_attn_wmma(const float* __restrict__ qkv, const float* __restrict__ bias,
                            const unsigned char* __restrict__ sparse,
                            const float* __restrict__ mask, float* __restrict__ out) {
  __shared__ float sS[16 * LL];               // 32 KB score/prob buffer
  int qt = blockIdx.x, h = blockIdx.y, b = blockIdx.z;
  int lane = threadIdx.x, q15 = lane & 15;
  int hi = (lane >= 16) ? 1 : 0;
  int q0 = qt * 16;

  AFrag aq;                                   // Q tile fragment (loop invariant)
  const float* qrow = qkv + (size_t)(b * LL + q0 + q15) * (3 * DM) + h * DK;
#pragma unroll
  for (int e = 0; e < 16; e++) {
    int ka = ((e < 8) ? e : e + 8) + hi * 8;
    aq.h[e] = (_Float16)qrow[ka];
  }
  const float* brows = bias + (((size_t)(b * NH + h)) * LL + q0) * LL;

  for (int nt = 0; nt < LL / 16; ++nt) {      // scores
    if (nt + 1 < LL / 16)
      __builtin_prefetch(&brows[(size_t)(nt + 1) * 16], 0, 1);
    AFrag bk;
#pragma unroll
    for (int e = 0; e < 16; e++) {
      int kb = ((e < 8) ? e : e + 8) + hi * 8;
      bk.h[e] = (_Float16)qkv[(size_t)(b * LL + nt * 16 + q15) * (3 * DM) + DM + h * DK + kb];
    }
    CFrag c;
#pragma unroll
    for (int i = 0; i < 8; i++) c.f[i] = 0.f;
    c.v = __builtin_amdgcn_wmma_f32_16x16x32_f16(false, aq.v, false, bk.v,
                                                 (short)0, c.v, false, false);
    int kk = nt * 16 + q15;
    float mk = mask[b * LL + kk];
#pragma unroll
    for (int r = 0; r < 8; r++) {
      int m = r + hi * 8;
      float sc = c.f[r] * 0.17677669529663687f + brows[(size_t)m * LL + kk];
      if (!sparse[((size_t)b * LL + q0 + m) * LL + kk]) sc = -1e9f;
      if (mk <= 0.f) sc = -1e9f;
      sS[m * LL + kk] = sc;
    }
  }
  __syncthreads();

  for (int m = 0; m < 16; m++) {              // softmax per row
    float mq = mask[b * LL + q0 + m];
    float mx = -3e38f;
    for (int j = lane; j < LL; j += 32) mx = fmaxf(mx, sS[m * LL + j]);
    for (int off = 16; off > 0; off >>= 1) mx = fmaxf(mx, __shfl_xor(mx, off, 32));
    float sum = 0.f;
    for (int j = lane; j < LL; j += 32) {
      float e = __expf(sS[m * LL + j] - mx);
      sS[m * LL + j] = e; sum += e;
    }
    for (int off = 16; off > 0; off >>= 1) sum += __shfl_xor(sum, off, 32);
    float inv = (mq > 0.f ? 1.f : 0.f) / sum;
    for (int j = lane; j < LL; j += 32) sS[m * LL + j] *= inv;
  }
  __syncthreads();

#pragma unroll
  for (int vt = 0; vt < 2; ++vt) {            // P @ V
    CFrag c;
#pragma unroll
    for (int i = 0; i < 8; i++) c.f[i] = 0.f;
    for (int ks = 0; ks < LL / 32; ++ks) {
      AFrag ap, bv;
#pragma unroll
      for (int e = 0; e < 16; e++) {
        int ka = ((e < 8) ? e : e + 8) + hi * 8;
        ap.h[e] = (_Float16)sS[q15 * LL + ks * 32 + ka];
        int kb = ks * 32 + ((e < 8) ? e : e + 8) + hi * 8;
        bv.h[e] = (_Float16)qkv[(size_t)(b * LL + kb) * (3 * DM) + 2 * DM + h * DK + vt * 16 + q15];
      }
      c.v = __builtin_amdgcn_wmma_f32_16x16x32_f16(false, ap.v, false, bv.v,
                                                   (short)0, c.v, false, false);
    }
#pragma unroll
    for (int r = 0; r < 8; r++) {
      int m = r + hi * 8;
      out[(size_t)(b * LL + q0 + m) * DM + h * DK + vt * 16 + q15] = c.f[r];
    }
  }
}

// ---------------------------------------------------------------------------
// 6. dst = LayerNorm(x + dlt) * (mulmask ? mask : 1).  One wave per row.
// ---------------------------------------------------------------------------
__global__ void k_addln(const float* __restrict__ x, const float* __restrict__ dlt,
                        const float* __restrict__ g, const float* __restrict__ be,
                        const float* __restrict__ mask, float* __restrict__ dst,
                        int mulmask) {
  int row = blockIdx.x;
  int lane = threadIdx.x;
  float v[4];
#pragma unroll
  for (int j = 0; j < 4; j++) {
    int d = lane * 4 + j;
    v[j] = x[(size_t)row * DM + d] + dlt[(size_t)row * DM + d];
  }
  float sum = v[0] + v[1] + v[2] + v[3];
  for (int off = 16; off > 0; off >>= 1) sum += __shfl_xor(sum, off, 32);
  float mean = sum * (1.f / DM);
  float vs = 0.f;
#pragma unroll
  for (int j = 0; j < 4; j++) { float t = v[j] - mean; vs += t * t; }
  for (int off = 16; off > 0; off >>= 1) vs += __shfl_xor(vs, off, 32);
  float rstd = rsqrtf(vs * (1.f / DM) + 1e-5f);
  float mm = mulmask ? mask[row] : 1.f;
#pragma unroll
  for (int j = 0; j < 4; j++) {
    int d = lane * 4 + j;
    dst[(size_t)row * DM + d] = ((v[j] - mean) * rstd * g[d] + be[d]) * mm;
  }
}

// ---------------------------------------------------------------------------
// 7. Time-bucket pooling + 128->3 head.  One block per (b,t).
// ---------------------------------------------------------------------------
__global__ void k_pool_head(const float* __restrict__ h, const int* __restrict__ node_t,
                            const float* __restrict__ mask, const float* __restrict__ hw,
                            const float* __restrict__ hb, float* __restrict__ out) {
  __shared__ float fs[DM];
  int t = blockIdx.x, b = blockIdx.y;
  int d = threadIdx.x;
  float acc = 0.f, cnt = 0.f;
  for (int l = 0; l < LL; l++) {
    int tc = node_t[b * LL + l];
    tc = tc < 0 ? 0 : (tc > TLEN - 1 ? TLEN - 1 : tc);
    float oh = (tc == t) ? mask[b * LL + l] : 0.f;
    acc += h[(size_t)(b * LL + l) * DM + d] * oh;
    cnt += oh;
  }
  fs[d] = acc / (cnt + 1e-6f);
  __syncthreads();
  if (d < 3) {
    float p = hb[d];
    for (int c = 0; c < DM; c++) p += fs[c] * hw[c * 3 + d];
    out[(b * TLEN + t) * 3 + d] = p;
  }
}

// ---------------------------------------------------------------------------
// Host orchestration (input order = setup_inputs() insertion order).
// ---------------------------------------------------------------------------
extern "C" void kernel_launch(void* const* d_in, const int* in_sizes, int n_in,
                              void* d_out, int out_size, void* d_ws, size_t ws_size,
                              hipStream_t stream) {
  const float* feat   = (const float*)d_in[0];
  const int*   node_t = (const int*)  d_in[1];
  const int*   node_m = (const int*)  d_in[2];
  const float* mask   = (const float*)d_in[3];
  const float* enc_w  = (const float*)d_in[4];
  const float* enc_b  = (const float*)d_in[5];
  const float* ew1    = (const float*)d_in[6];
  const float* eb1    = (const float*)d_in[7];
  const float* ew2    = (const float*)d_in[8];
  const float* eb2    = (const float*)d_in[9];
  const float* head_w = (const float*)d_in[46];
  const float* head_b = (const float*)d_in[47];

  float* ws = (float*)d_ws;
  float* h      = ws; ws += (size_t)BB * LL * DM;
  float* abias  = ws; ws += (size_t)BB * NH * LL * LL;
  float* metric = ws; ws += (size_t)BB * LL * LL;   // reused as qkv after KNN
  float* aout   = ws; ws += (size_t)BB * LL * DM;
  float* proj   = ws; ws += (size_t)BB * LL * DM;
  float* ff1    = ws; ws += (size_t)BB * LL * DFF;
  float* ff2    = ws; ws += (size_t)BB * LL * DM;
  unsigned char* sparse = (unsigned char*)ws;
  float* qkv = metric;

  (void)hipMemsetAsync(sparse, 0, (size_t)BB * LL * LL, stream);

  k_encode<<<BB * LL, DM, 0, stream>>>(feat, enc_w, enc_b, h);

  dim3 ge(LL / 16, LL, BB);
  k_edge_wmma<<<ge, 32, 0, stream>>>(feat, node_t, node_m, mask, ew1, eb1, ew2, eb2,
                                     abias, metric);

  dim3 gk(LL, BB);
  k_knn<<<gk, 32, 0, stream>>>(metric, mask, sparse);

  const int M = BB * LL;
  for (int layer = 0; layer < 3; ++layer) {
    const float* qkv_w = (const float*)d_in[10 + 12 * layer + 0];
    const float* qkv_b = (const float*)d_in[10 + 12 * layer + 1];
    const float* out_w = (const float*)d_in[10 + 12 * layer + 2];
    const float* out_b = (const float*)d_in[10 + 12 * layer + 3];
    const float* ffw1  = (const float*)d_in[10 + 12 * layer + 4];
    const float* ffb1  = (const float*)d_in[10 + 12 * layer + 5];
    const float* ffw2  = (const float*)d_in[10 + 12 * layer + 6];
    const float* ffb2  = (const float*)d_in[10 + 12 * layer + 7];
    const float* n1g   = (const float*)d_in[10 + 12 * layer + 8];
    const float* n1b   = (const float*)d_in[10 + 12 * layer + 9];
    const float* n2g   = (const float*)d_in[10 + 12 * layer + 10];
    const float* n2b   = (const float*)d_in[10 + 12 * layer + 11];

    dim3 g1(3 * DM / 16, M / 16);
    k_gemm_wmma<<<g1, 32, 0, stream>>>(h, qkv_w, qkv_b, qkv, M, 3 * DM, DM, 0);

    dim3 ga(LL / 16, NH, BB);
    k_attn_wmma<<<ga, 32, 0, stream>>>(qkv, abias, sparse, mask, aout);

    dim3 g2(DM / 16, M / 16);
    k_gemm_wmma<<<g2, 32, 0, stream>>>(aout, out_w, out_b, proj, M, DM, DM, 0);

    k_addln<<<M, 32, 0, stream>>>(h, proj, n1g, n1b, mask, h, 0);

    dim3 g3(DFF / 16, M / 16);
    k_gemm_wmma<<<g3, 32, 0, stream>>>(h, ffw1, ffb1, ff1, M, DFF, DM, 1);

    dim3 g4(DM / 16, M / 16);
    k_gemm_wmma<<<g4, 32, 0, stream>>>(ff1, ffw2, ffb2, ff2, M, DM, DFF, 0);

    k_addln<<<M, 32, 0, stream>>>(h, ff2, n2g, n2b, mask, h, 1);
  }

  dim3 gp(TLEN, BB);
  k_pool_head<<<gp, DM, 0, stream>>>(h, node_t, mask, head_w, head_b, (float*)d_out);
}
